// VectorQuantizer_34187939676277
// MI455X (gfx1250) — matrix operand order repaired and checked by reading, not compiled, not run on hardware
//
#include <hip/hip_runtime.h>

#define B_N 32768
#define K_N 8192
#define D_N 512
#define EMA_DECAY 0.99f
#define ONE_MINUS 0.01f
#define COMMIT 0.25f
#define EPSV 1e-5f

typedef __attribute__((ext_vector_type(16))) __bf16 v16bf;
typedef __attribute__((ext_vector_type(8))) float v8f;

struct alignas(16) Q4 { unsigned a, b, c, d; };
union Frag { v16bf v; Q4 q[2]; };

__device__ __forceinline__ unsigned short f2bf(float f) {
  unsigned u = __float_as_uint(f);
  u += 0x7fffu + ((u >> 16) & 1u);   // RNE
  return (unsigned short)(u >> 16);
}
__device__ __forceinline__ float bf2f(unsigned short h) {
  return __uint_as_float(((unsigned)h) << 16);
}
__device__ __forceinline__ unsigned pack2(unsigned short lo, unsigned short hi) {
  return (unsigned)lo | ((unsigned)hi << 16);
}

// ---- LDS layout (bytes), fragment-major (16B granules = exactly one b128 fragment) ----
// A plane per wave: granule(kk,q,lane) at kk*1024 + q*512 + lane*16   (16KB/wave)
// B tile:           granule(kk,h,q,n)  at kk*1024 + h*512 + q*256 + n*16 (16KB/plane)
#define ZHI_OFF 0u
#define ZLO_OFF 131072u            // 8 waves * 16KB
#define CT_OFF  262144u            // two buffers of (hi 16KB + lo 16KB)
#define CT_LO   16384u
#define CT_BUFSZ 32768u
#define LDS_BYTES 327680u          // == 320KB WGP LDS, exact fit

// ---------------- kernel 0a: init accumulators, pre-scale ema_weight ----------------
__global__ void __launch_bounds__(256) vq_init(const float* __restrict__ emaW,
                                               float* __restrict__ o_w,
                                               float* __restrict__ counts,
                                               float* __restrict__ lossSum,
                                               float* __restrict__ nSum) {
  size_t i = (size_t)blockIdx.x * 256 + threadIdx.x;
  o_w[i] = EMA_DECAY * emaW[i];            // atomic scatter adds 0.01*z later
  if (i < K_N) counts[i] = 0.f;
  if (i == 0) { lossSum[0] = 0.f; nSum[0] = 0.f; }
}

// ---------------- kernel 0b: codebook row norms ----------------
__global__ void __launch_bounds__(256) vq_cnorm(const float* __restrict__ codebook,
                                                float* __restrict__ cnorm) {
  int wave = threadIdx.x >> 5, lane = threadIdx.x & 31;
  int k = blockIdx.x * 8 + wave;
  const float* cr = codebook + (size_t)k * D_N;
  float s = 0.f;
  for (int j = lane; j < D_N; j += 32) { float v = cr[j]; s = fmaf(v, v, s); }
#pragma unroll
  for (int m = 16; m >= 1; m >>= 1) s += __shfl_xor(s, m, 32);
  if (lane == 0) cnorm[k] = s;
}

// ---------------- kernel 0c: one-time codebook split into bf16 hi/lo planes ----------------
// Planes live in the (not-yet-needed) z_q output region: 16MB of 64MB, 16B-aligned.
__global__ void __launch_bounds__(256) vq_split_codebook(const float* __restrict__ codebook,
                                                         unsigned short* __restrict__ hiP,
                                                         unsigned short* __restrict__ loP) {
  size_t i = (size_t)blockIdx.x * 256 + threadIdx.x;   // float4 index
  float4 v = reinterpret_cast<const float4*>(codebook)[i];
  unsigned short h0 = f2bf(v.x), h1 = f2bf(v.y), h2 = f2bf(v.z), h3 = f2bf(v.w);
  unsigned short l0 = f2bf(v.x - bf2f(h0)), l1 = f2bf(v.y - bf2f(h1));
  unsigned short l2 = f2bf(v.z - bf2f(h2)), l3 = f2bf(v.w - bf2f(h3));
  size_t e = i * 4;
  unsigned* ph = reinterpret_cast<unsigned*>(hiP + e);
  ph[0] = pack2(h0, h1); ph[1] = pack2(h2, h3);
  unsigned* pl = reinterpret_cast<unsigned*>(loP + e);
  pl[0] = pack2(l0, l1); pl[1] = pack2(l2, l3);
}

// ---------------- kernel 1: fused distance GEMM + argmin (split-bf16 WMMA) ----------------
__global__ void __launch_bounds__(256, 2) vq_dist_argmin(const float* __restrict__ z_e,
                                                         const unsigned short* __restrict__ hiP,
                                                         const unsigned short* __restrict__ loP,
                                                         const float* __restrict__ cnorm,
                                                         int* __restrict__ idxI,
                                                         float* __restrict__ idxF) {
  extern __shared__ char smem[];
  const int tid = threadIdx.x;
  const int wave = tid >> 5, lane = tid & 31;
  const int h = lane >> 4, ln = lane & 15;
  const int rowBase = blockIdx.x * 128;

  // Per-thread async-copy descriptors: 8 granules of the 16x512 code tile
  // (j=0..3 hi plane, j=4..7 lo plane). dstA starts at buffer 0.
  unsigned dstA[8], offA[8];
#pragma unroll
  for (int j = 0; j < 8; ++j) {
    unsigned u = (unsigned)tid + (unsigned)(j & 3) * 256u;     // Q4 index in tile plane
    unsigned kk = (u >> 2) & 15u, hh = u & 1u, qq = (u >> 1) & 1u, n = u >> 6;
    unsigned slot = kk * 1024u + hh * 512u + qq * 256u + n * 16u + ((j >= 4) ? CT_LO : 0u);
    dstA[j] = CT_OFF + slot;
    offA[j] = ((j >= 4) ? 8388608u : 0u) + u * 16u;            // lo plane is +8MB in global
  }
  const unsigned long long gbase = (unsigned long long)(uintptr_t)hiP;

  // Prologue: async-copy tile 0 into buffer 0 (overlaps with z staging below).
#pragma unroll
  for (int j = 0; j < 8; ++j)
    asm volatile("global_load_async_to_lds_b128 %0, %1, %2"
                 :: "v"(dstA[j]), "v"(offA[j]), "s"(gbase) : "memory");

  // Stage 128 z rows as split-bf16, fragment-major (converted once per block).
  {
    const float4* zb = reinterpret_cast<const float4*>(z_e + (size_t)rowBase * D_N);
    for (int i = tid; i < (128 * D_N) / 4; i += 256) {
      int e = i * 4, row = e >> 9, col = e & 511;
      unsigned wv = (unsigned)row >> 4, m = (unsigned)row & 15u;
      unsigned kk = (unsigned)col >> 5, r = (unsigned)col & 31u;
      unsigned hh = (r >> 3) & 1u, qq = (r >> 4) & 1u, pos = (unsigned)col & 7u;
      unsigned off = wv * 16384u + kk * 1024u + qq * 512u + (m + 16u * hh) * 16u + pos * 2u;
      float4 v = zb[i];
      unsigned short h0 = f2bf(v.x), h1 = f2bf(v.y), h2 = f2bf(v.z), h3 = f2bf(v.w);
      unsigned short l0 = f2bf(v.x - bf2f(h0)), l1 = f2bf(v.y - bf2f(h1));
      unsigned short l2 = f2bf(v.z - bf2f(h2)), l3 = f2bf(v.w - bf2f(h3));
      unsigned* p = reinterpret_cast<unsigned*>(smem + ZHI_OFF + off);
      p[0] = pack2(h0, h1); p[1] = pack2(h2, h3);
      unsigned* q = reinterpret_cast<unsigned*>(smem + ZLO_OFF + off);
      q[0] = pack2(l0, l1); q[1] = pack2(l2, l3);
    }
  }

  unsigned aHiB = ZHI_OFF + (unsigned)wave * 16384u + (unsigned)lane * 16u;
  unsigned aLoB = ZLO_OFF + (unsigned)wave * 16384u + (unsigned)lane * 16u;
  asm("" : "+v"(aHiB));  // pin bases in VGPRs so per-kk deltas stay DS immediates
  asm("" : "+v"(aLoB));
  const unsigned bLaneOff = (unsigned)h * 512u + (unsigned)ln * 16u;

  float best[8]; int bidx[8];
#pragma unroll
  for (int r = 0; r < 8; ++r) { best[r] = 3.4e38f; bidx[r] = 0; }

  for (int t = 0; t < K_N / 16; ++t) {
    asm volatile("s_wait_asynccnt 0x0" ::: "memory");  // own tile-t copies landed
    __syncthreads();                                   // all waves: tile t ready, buf[nxt] free
    if (t + 1 < K_N / 16) {                            // async-copy tile t+1 into other buffer
#pragma unroll
      for (int j = 0; j < 8; ++j) dstA[j] ^= CT_BUFSZ;
      const unsigned long long gb = gbase + (unsigned long long)(t + 1) * 16384ull;
#pragma unroll
      for (int j = 0; j < 8; ++j)
        asm volatile("global_load_async_to_lds_b128 %0, %1, %2"
                     :: "v"(dstA[j]), "v"(offA[j]), "s"(gb) : "memory");
    }

    // Per-tile B base as one pinned 32-bit LDS index; all per-kk deltas
    // (kk*1024 + {0,256} + {0,CT_LO}) fit the 16-bit DS offset field.
    unsigned bB = CT_OFF + (unsigned)(t & 1) * CT_BUFSZ + bLaneOff;
    asm("" : "+v"(bB));

    v8f acc = {0.f, 0.f, 0.f, 0.f, 0.f, 0.f, 0.f, 0.f};
#pragma unroll
    for (int kk = 0; kk < 16; ++kk) {
      const unsigned o = (unsigned)kk * 1024u;
      Frag aH, aL, bH, bL;
      aH.q[0] = *reinterpret_cast<const Q4*>(smem + aHiB + o);
      aH.q[1] = *reinterpret_cast<const Q4*>(smem + aHiB + o + 512u);
      aL.q[0] = *reinterpret_cast<const Q4*>(smem + aLoB + o);
      aL.q[1] = *reinterpret_cast<const Q4*>(smem + aLoB + o + 512u);
      bH.q[0] = *reinterpret_cast<const Q4*>(smem + bB + o);
      bH.q[1] = *reinterpret_cast<const Q4*>(smem + bB + o + 256u);
      bL.q[0] = *reinterpret_cast<const Q4*>(smem + bB + CT_LO + o);
      bL.q[1] = *reinterpret_cast<const Q4*>(smem + bB + CT_LO + o + 256u);
      // split product: hi*hi + hi*lo + lo*hi  (~fp32-accurate dot)
      acc = __builtin_amdgcn_wmma_f32_16x16x32_bf16(false, aH.v, false, bH.v, (short)0, acc, false, false);
      acc = __builtin_amdgcn_wmma_f32_16x16x32_bf16(false, aH.v, false, bL.v, (short)0, acc, false, false);
      acc = __builtin_amdgcn_wmma_f32_16x16x32_bf16(false, aL.v, false, bH.v, (short)0, acc, false, false);
    }

    const int n0 = t * 16;
    const float cn = cnorm[n0 + ln];
    const int nIdx = n0 + ln;
#pragma unroll
    for (int r = 0; r < 8; ++r) {
      float s = fmaf(-2.f, acc[r], cn);  // ||z||^2 constant per row: irrelevant for argmin
      if (s < best[r]) { best[r] = s; bidx[r] = nIdx; }
    }
  }

  // Reduce across the 16 lanes of each C-matrix half (xor masks stay in-half).
#pragma unroll
  for (int r = 0; r < 8; ++r) {
    float bv = best[r]; int bi = bidx[r];
#pragma unroll
    for (int m = 8; m >= 1; m >>= 1) {
      float ov = __shfl_xor(bv, m, 32);
      int oi = __shfl_xor(bi, m, 32);
      if (ov < bv || (ov == bv && oi < bi)) { bv = ov; bi = oi; }
    }
    if (ln == 0) {
      int row = rowBase + wave * 16 + r + 8 * h;
      idxI[row] = bi;
      idxF[row] = (float)bi;
    }
  }
}

// ---------------- kernel 2: gather z_q, commitment loss, segment-sum scatter ----------------
__global__ void __launch_bounds__(256) vq_gather_scatter(const float* __restrict__ z_e,
                                                         const float* __restrict__ codebook,
                                                         const int* __restrict__ idxI,
                                                         float* __restrict__ o_zq,
                                                         float* __restrict__ o_w,
                                                         float* __restrict__ counts,
                                                         float* __restrict__ lossSum) {
  __shared__ float warpAcc[8];
  const int b = blockIdx.x, tid = threadIdx.x;
  const int idx = idxI[b];
  const float* zr = z_e + (size_t)b * D_N;
  const float* cr = codebook + (size_t)idx * D_N;
  float* wq = o_w + (size_t)idx * D_N;
  float* zq = o_zq + (size_t)b * D_N;
  float ls = 0.f;
#pragma unroll
  for (int t = 0; t < 2; ++t) {
    int d = tid + t * 256;
    float c = cr[d], z = zr[d];
    zq[d] = c;                               // z_q_st == z_q numerically
    float df = z - c;
    ls = fmaf(df, df, ls);
    atomicAdd(&wq[d], ONE_MINUS * z);        // (1-decay) * segment_sum(z)
  }
#pragma unroll
  for (int m = 16; m >= 1; m >>= 1) ls += __shfl_xor(ls, m, 32);
  if ((tid & 31) == 0) warpAcc[tid >> 5] = ls;
  __syncthreads();
  if (tid == 0) {
    float s = 0.f;
#pragma unroll
    for (int w = 0; w < 8; ++w) s += warpAcc[w];
    atomicAdd(lossSum, s);
    atomicAdd(&counts[idx], 1.0f);
  }
}

// ---------------- kernel 3a: count EMA, n reduction, loss finalize ----------------
__global__ void __launch_bounds__(256) vq_count_ema(const float* __restrict__ emaC,
                                                    const float* __restrict__ counts,
                                                    float* __restrict__ o_cnt,
                                                    float* __restrict__ nSum,
                                                    const float* __restrict__ lossSum,
                                                    float* __restrict__ o_loss) {
  int i = blockIdx.x * 256 + threadIdx.x;
  float nc = EMA_DECAY * emaC[i] + ONE_MINUS * counts[i];
  o_cnt[i] = nc;
  float s = nc;
#pragma unroll
  for (int m = 16; m >= 1; m >>= 1) s += __shfl_xor(s, m, 32);
  if ((threadIdx.x & 31) == 0) atomicAdd(nSum, s);
  if (i == 0) o_loss[0] = COMMIT * lossSum[0] / 16777216.f;  // mean over B*D
}

// ---------------- kernel 3b: smoothed counts -> new codebook ----------------
__global__ void __launch_bounds__(256) vq_codebook_out(const float* __restrict__ o_w,
                                                       const float* __restrict__ o_cnt,
                                                       const float* __restrict__ nSum,
                                                       float* __restrict__ o_cb) {
  size_t i = (size_t)blockIdx.x * 256 + threadIdx.x;
  int k = (int)(i >> 9);
  float n = nSum[0];
  float sm = (o_cnt[k] + EPSV) / (n + (float)K_N * EPSV) * n;
  o_cb[i] = o_w[i] / sm;
}

extern "C" void kernel_launch(void* const* d_in, const int* in_sizes, int n_in,
                              void* d_out, int out_size, void* d_ws, size_t ws_size,
                              hipStream_t stream) {
  const float* z_e  = (const float*)d_in[0];
  const float* cb   = (const float*)d_in[1];
  const float* emaC = (const float*)d_in[2];
  const float* emaW = (const float*)d_in[3];
  float* out = (float*)d_out;
  float* ws  = (float*)d_ws;

  const size_t ND = (size_t)B_N * D_N;   // 16777216
  const size_t NK = (size_t)K_N * D_N;   // 4194304
  float* o_zq   = out;                   // [B,D]
  float* o_idx  = out + ND;              // [B] indices as float
  float* o_loss = out + ND + B_N;        // [1]
  float* o_cb   = out + ND + B_N + 1;    // [K,D]
  float* o_cnt  = o_cb + NK;             // [K]
  float* o_w    = o_cnt + K_N;           // [K,D] new_ema_weight (also accumulator)

  float* cnorm   = ws;                   // [K]
  float* counts  = ws + K_N;             // [K]
  float* lossSum = ws + 2 * K_N;         // [1]
  float* nSum    = ws + 2 * K_N + 1;     // [1]
  int*   idxI    = (int*)(ws + 2 * K_N + 8);  // [B]

  // Borrow the z_q output region (16B-aligned, 64MB) for the split codebook planes;
  // vq_gather_scatter fully overwrites it afterwards.
  unsigned short* hiP = (unsigned short*)out;      // K*D bf16 (8MB)
  unsigned short* loP = hiP + NK;                  // K*D bf16 (8MB)

  vq_init<<<(unsigned)(NK / 256), 256, 0, stream>>>(emaW, o_w, counts, lossSum, nSum);
  vq_cnorm<<<K_N / 8, 256, 0, stream>>>(cb, cnorm);
  vq_split_codebook<<<(unsigned)(NK / 4 / 256), 256, 0, stream>>>(cb, hiP, loP);
  vq_dist_argmin<<<B_N / 128, 256, LDS_BYTES, stream>>>(z_e, hiP, loP, cnorm, idxI, o_idx);
  vq_gather_scatter<<<B_N, 256, 0, stream>>>(z_e, cb, idxI, o_zq, o_w, counts, lossSum);
  vq_count_ema<<<K_N / 256, 256, 0, stream>>>(emaC, counts, o_cnt, nSum, lossSum, o_loss);
  vq_codebook_out<<<(unsigned)(NK / 256), 256, 0, stream>>>(o_w, o_cnt, nSum, o_cb);
}
